// Voc2Voc_81913616269325
// MI455X (gfx1250) — matrix operand order repaired and checked by reading, not compiled, not run on hardware
//
#include <hip/hip_runtime.h>
#include <hip/hip_bf16.h>

// ---------------- types ----------------
typedef __attribute__((ext_vector_type(16))) __bf16        v16bf;
typedef __attribute__((ext_vector_type(8)))  float         v8f;
typedef __attribute__((ext_vector_type(8)))  unsigned int  v8u;
typedef __attribute__((ext_vector_type(4)))  int           v4i;

// ---------------- gfx1250 async global->LDS path (guarded) ----------------
#if defined(__HIP_DEVICE_COMPILE__) && __has_builtin(__builtin_amdgcn_global_load_async_to_lds_b128)
#define HAVE_ASYNC 1
typedef __attribute__((address_space(1))) v4i gv4i;   // global <4 x i32>
typedef __attribute__((address_space(3))) v4i lv4i;   // LDS    <4 x i32>
// copies 16 bytes per lane directly into LDS, tracked by ASYNCcnt
#define COPY16(g, l, OFF)                                                    \
    __builtin_amdgcn_global_load_async_to_lds_b128(                          \
        (gv4i*)(void*)(g), (lv4i*)(void*)(l), (OFF), 0)
#else
#define HAVE_ASYNC 0
#define COPY16(g, l, OFF)                                                    \
    (*reinterpret_cast<uint4*>((__bf16*)(l) + (OFF) / 2) =                   \
     *reinterpret_cast<const uint4*>((const __bf16*)(g) + (OFF) / 2))
#endif

static __device__ inline void wait_async_zero() {
#if HAVE_ASYNC
#if __has_builtin(__builtin_amdgcn_s_wait_asynccnt)
    __builtin_amdgcn_s_wait_asynccnt(0);
#else
    asm volatile("s_wait_asynccnt 0x0" ::: "memory");
#endif
#endif
}

static __device__ inline unsigned int pack_bf16x2(float a, float b) {
    unsigned short lo = __builtin_bit_cast(unsigned short, (__bf16)a);
    unsigned short hi = __builtin_bit_cast(unsigned short, (__bf16)b);
    return (unsigned int)lo | ((unsigned int)hi << 16);
}

// Load a 16-element bf16 fragment as two 16B LDS reads (addresses 16B-aligned).
static __device__ inline v16bf lds_frag(const __bf16* p0, const __bf16* p1) {
    const uint4 a = *reinterpret_cast<const uint4*>(p0);
    const uint4 b = *reinterpret_cast<const uint4*>(p1);
    v8u r;
    r[0] = a.x; r[1] = a.y; r[2] = a.z; r[3] = a.w;
    r[4] = b.x; r[5] = b.y; r[6] = b.z; r[7] = b.w;
    return __builtin_bit_cast(v16bf, r);
}

// ---------------- prep kernels ----------------

// out[row,:] = emb[idx[row],:] * 32 (sqrt(1024)), cast to bf16. H = 1024, block 256.
__global__ void gather_scale_bf16(const int* __restrict__ idx,
                                  const float* __restrict__ emb,
                                  __bf16* __restrict__ out, int H) {
    const int row = blockIdx.x;
    const int c   = threadIdx.x * 4;
    const int id  = idx[row];
    const float4 f = *reinterpret_cast<const float4*>(emb + (size_t)id * H + c);
    uint2 r;
    r.x = pack_bf16x2(f.x * 32.0f, f.y * 32.0f);
    r.y = pack_bf16x2(f.z * 32.0f, f.w * 32.0f);
    *reinterpret_cast<uint2*>(out + (size_t)row * H + c) = r;
}

// f32 src[R][C] -> bf16 dst[C][R], 32x32 LDS tiles, block (32,8).
__global__ void transpose_cast_w(const float* __restrict__ src,
                                 __bf16* __restrict__ dst, int R, int C) {
    __shared__ float tile[32][33];
    const int c0 = blockIdx.x * 32, r0 = blockIdx.y * 32;
    const int tx = threadIdx.x, ty = threadIdx.y;
#pragma unroll
    for (int i = 0; i < 4; ++i)
        tile[ty * 4 + i][tx] = src[(size_t)(r0 + ty * 4 + i) * C + c0 + tx];
    __syncthreads();
#pragma unroll
    for (int i = 0; i < 4; ++i)
        dst[(size_t)(c0 + ty * 4 + i) * R + r0 + tx] = (__bf16)tile[tx][ty * 4 + i];
}

// bf16 src[z][R][C] -> bf16 dst[z][C][R], block (32,8), grid (C/32, R/32, Z).
__global__ void transpose_bf16(const __bf16* __restrict__ src,
                               __bf16* __restrict__ dst, int R, int C) {
    __shared__ unsigned short tile[32][34];
    const size_t base = (size_t)blockIdx.z * R * C;
    const int c0 = blockIdx.x * 32, r0 = blockIdx.y * 32;
    const int tx = threadIdx.x, ty = threadIdx.y;
#pragma unroll
    for (int i = 0; i < 4; ++i)
        tile[ty * 4 + i][tx] = __builtin_bit_cast(
            unsigned short, src[base + (size_t)(r0 + ty * 4 + i) * C + c0 + tx]);
    __syncthreads();
#pragma unroll
    for (int i = 0; i < 4; ++i)
        dst[base + (size_t)(c0 + ty * 4 + i) * R + r0 + tx] =
            __builtin_bit_cast(__bf16, tile[tx][ty * 4 + i]);
}

// ---------------- softmax (in-place: f32 row -> bf16 weights) ----------------
__global__ void softmax_rows(float* __restrict__ logits) {
    __shared__ float red[256];
    const int tid = threadIdx.x;
    float* rp = logits + (size_t)blockIdx.x * 2048;

    float v[8];
#pragma unroll
    for (int i = 0; i < 8; ++i) v[i] = rp[i * 256 + tid];   // all global reads first

    float mx = v[0];
#pragma unroll
    for (int i = 1; i < 8; ++i) mx = fmaxf(mx, v[i]);
    red[tid] = mx; __syncthreads();
    for (int s = 128; s > 0; s >>= 1) {
        if (tid < s) red[tid] = fmaxf(red[tid], red[tid + s]);
        __syncthreads();
    }
    mx = red[0]; __syncthreads();

    float e[8], sum = 0.0f;
#pragma unroll
    for (int i = 0; i < 8; ++i) { e[i] = __expf(v[i] - mx); sum += e[i]; }
    red[tid] = sum; __syncthreads();
    for (int s = 128; s > 0; s >>= 1) {
        if (tid < s) red[tid] += red[tid + s];
        __syncthreads();
    }
    const float inv = 1.0f / red[0];

    __bf16* wp = reinterpret_cast<__bf16*>(rp);
#pragma unroll
    for (int i = 0; i < 8; ++i) wp[i * 256 + tid] = (__bf16)(e[i] * inv);
}

// ---------------- WMMA GEMM (NT only: B is [N][K] row-major) ----------------
// C[M,N] = A[M,K] * B^T, bf16 in / f32 accumulate via v_wmma_f32_16x16x32_bf16.
// IM2COL=1: A is attn bf16 [4*2048, 1024]; A[gm,k] = attn[b, t + k/1024 - 1, k%1024].
// MODE 0: C f32 = alpha*acc          MODE 1: C bf16 = acc
// MODE 2: C f32 = acc + bias[n]      MODE 3: C bf16 = relu(acc + bias[n])
// Block 256 (8 waves as 2x4), tile 64x256, BK=32, double-buffered async staging.
// All problem dims divide the tiles exactly -> full EXEC (WMMA requirement).
template <int MODE, bool IM2COL>
__global__ __launch_bounds__(256) void gemm_wmma(
    const __bf16* __restrict__ A, int lda,
    const __bf16* __restrict__ Bm, int ldb,
    void* __restrict__ Cc, int ldc,
    int K, float alpha, const float* __restrict__ bias) {

    __shared__ __bf16 As[2][64 * 32];
    __shared__ __bf16 Bs[2][256 * 32];

    const int tid    = threadIdx.x;
    const int gmBase = blockIdx.y * 64;
    const int gnBase = blockIdx.x * 256;
    const int lane   = tid & 31;
    const int wave   = tid >> 5;
    const int wm     = wave >> 2;    // 0..1
    const int wn     = wave & 3;     // 0..3

    v8f acc[2][4] = {};

    const int arow = tid >> 2;           // 0..63
    const int akc  = (tid & 3) * 8;      // 0,8,16,24
    const int brow = tid;                // 0..255

    auto stageA = [&](int kk, int buf) {
        __bf16* dstp = &As[buf][arow * 32 + akc];
        if constexpr (IM2COL) {
            const int gm = gmBase + arow;
            const int bb = gm >> 11, t = gm & 2047;
            const int kg = kk + akc, ks = kg >> 10, d = kg & 1023;
            const int tp = t + ks - 1;                     // SAME padding
            if ((unsigned)tp < 2048u) {
                COPY16(A + ((size_t)bb * 2048 + tp) * lda + d, dstp, 0);
            } else {
                *reinterpret_cast<uint4*>(dstp) = make_uint4(0u, 0u, 0u, 0u);
            }
        } else {
            COPY16(A + (size_t)(gmBase + arow) * lda + kk + akc, dstp, 0);
        }
    };
    auto stageB = [&](int kk, int buf) {
        const __bf16* srcp = Bm + (size_t)(gnBase + brow) * ldb + kk;
        __bf16* dstp = &Bs[buf][brow * 32];
        COPY16(srcp, dstp, 0);
        COPY16(srcp, dstp, 16);
    };

    const int steps = K >> 5;
    stageA(0, 0);
    stageB(0, 0);

    const int mr    = lane & 15;
    const int koffA = (lane >> 4) * 8;   // A frag: K = koff..+7, koff+16..+23
    const int kbB   = (lane >> 4) * 8;   // B frag: K = koff..+7, koff+16..+23

    for (int i = 0; i < steps; ++i) {
        const int cur = i & 1;
        wait_async_zero();
        __syncthreads();                  // all staging for buf[cur] visible; all
                                          // waves done reading buf[cur^1]
        if (i + 1 < steps) {
            stageA((i + 1) * 32, cur ^ 1);
            stageB((i + 1) * 32, cur ^ 1);
        }

        v16bf afr[2], bfr[4];
#pragma unroll
        for (int mt = 0; mt < 2; ++mt) {
            const __bf16* p = &As[cur][(wm * 32 + mt * 16 + mr) * 32 + koffA];
            afr[mt] = lds_frag(p, p + 16);
        }
#pragma unroll
        for (int nt = 0; nt < 4; ++nt) {
            const __bf16* p = &Bs[cur][(wn * 64 + nt * 16 + mr) * 32 + kbB];
            bfr[nt] = lds_frag(p, p + 16);
        }
#pragma unroll
        for (int mt = 0; mt < 2; ++mt)
#pragma unroll
            for (int nt = 0; nt < 4; ++nt)
                acc[mt][nt] = __builtin_amdgcn_wmma_f32_16x16x32_bf16(
                    false, afr[mt], false, bfr[nt],
                    (short)0, acc[mt][nt], false, false);
    }

    // ---- epilogue ----
    const int rhalf = (lane >> 4) * 8;
    const int cn    = lane & 15;
#pragma unroll
    for (int mt = 0; mt < 2; ++mt)
#pragma unroll
        for (int nt = 0; nt < 4; ++nt)
#pragma unroll
            for (int r = 0; r < 8; ++r) {
                const int gm = gmBase + wm * 32 + mt * 16 + rhalf + r;
                const int gn = gnBase + wn * 64 + nt * 16 + cn;
                const float v = acc[mt][nt][r];
                if constexpr (MODE == 0) {
                    ((float*)Cc)[(size_t)gm * ldc + gn] = v * alpha;
                } else if constexpr (MODE == 1) {
                    ((__bf16*)Cc)[(size_t)gm * ldc + gn] = (__bf16)v;
                } else if constexpr (MODE == 2) {
                    ((float*)Cc)[(size_t)gm * ldc + gn] = v + bias[gn];
                } else {
                    const float t = v + bias[gn];
                    ((__bf16*)Cc)[(size_t)gm * ldc + gn] = (__bf16)fmaxf(t, 0.0f);
                }
            }
    (void)alpha;
}

// ---------------- host launcher ----------------
extern "C" void kernel_launch(void* const* d_in, const int* in_sizes, int n_in,
                              void* d_out, int out_size, void* d_ws, size_t ws_size,
                              hipStream_t stream) {
    (void)in_sizes; (void)n_in; (void)out_size; (void)ws_size;

    const int*   inputs     = (const int*)  d_in[0];  // [4,2048]
    const int*   targets    = (const int*)  d_in[1];  // [4,2048]
    const float* input_emb  = (const float*)d_in[2];  // [32000,1024]
    const float* target_emb = (const float*)d_in[3];  // [32000,1024]
    const float* conv_w     = (const float*)d_in[4];  // [3,1024,4096]
    const float* conv_b     = (const float*)d_in[5];  // [4096]
    const float* dense_w    = (const float*)d_in[6];  // [4096,1024]
    const float* dense_b    = (const float*)d_in[7];  // [1024]
    float*       out        = (float*)d_out;          // [4,2048,1024] f32

    constexpr int Bn = 4, T = 2048, Hh = 1024, U = 4096;
    constexpr size_t MB = 1024 * 1024;

    // workspace layout (176 MiB total)
    char* ws = (char*)d_ws;
    __bf16* te     = (__bf16*)(ws + 0 * MB);    // 16 MiB [8192,1024]
    __bf16* ie     = (__bf16*)(ws + 16 * MB);   // 16 MiB [8192,1024]
    __bf16* ieT    = (__bf16*)(ws + 32 * MB);   // 16 MiB [4][1024][2048]
    __bf16* cwT    = (__bf16*)(ws + 48 * MB);   // 24 MiB [4096,3072]
    __bf16* dwT    = (__bf16*)(ws + 72 * MB);   //  8 MiB [1024,4096]
    float*  logits = (float*) (ws + 80 * MB);   // 16 MiB [2048,2048] per-batch reuse
    __bf16* attn   = (__bf16*)(ws + 96 * MB);   // 16 MiB [8192,1024]
    __bf16* hbuf   = (__bf16*)(ws + 112 * MB);  // 64 MiB [8192,4096]

    // 1) embedding gathers (+ sqrt(H)=32 scale) -> bf16
    gather_scale_bf16<<<Bn * T, 256, 0, stream>>>(targets, target_emb, te, Hh);
    gather_scale_bf16<<<Bn * T, 256, 0, stream>>>(inputs,  input_emb,  ie, Hh);

    // 2) B-operand prep: transposed bf16 copies so every GEMM is NT / async-friendly
    transpose_bf16<<<dim3(Hh / 32, T / 32, Bn), dim3(32, 8), 0, stream>>>(ie, ieT, T, Hh);
    transpose_cast_w<<<dim3(U / 32, 3 * Hh / 32), dim3(32, 8), 0, stream>>>(
        conv_w, cwT, 3 * Hh, U);                     // [3072,4096] -> [4096,3072]
    transpose_cast_w<<<dim3(Hh / 32, U / 32), dim3(32, 8), 0, stream>>>(
        dense_w, dwT, U, Hh);                        // [4096,1024] -> [1024,4096]

    // 3) attention per batch (stream-serialized so the logits buffer is reused)
    for (int b = 0; b < Bn; ++b) {
        const __bf16* teb  = te  + (size_t)b * T * Hh;
        const __bf16* ieb  = ie  + (size_t)b * T * Hh;
        const __bf16* ieTb = ieT + (size_t)b * T * Hh;
        // logits[T,T] = (te * ie^T) / sqrt(H)   (embeddings already carry x32 each)
        gemm_wmma<0, false><<<dim3(T / 256, T / 64), 256, 0, stream>>>(
            teb, Hh, ieb, Hh, logits, T, Hh, 1.0f / 32.0f, nullptr);
        softmax_rows<<<T, 256, 0, stream>>>(logits);
        // attn[T,H] = weights * ie   (B = ieT [H][T], weights row stride 2T bf16)
        gemm_wmma<1, false><<<dim3(Hh / 256, T / 64), 256, 0, stream>>>(
            (const __bf16*)logits, 2 * T, ieTb, T,
            attn + (size_t)b * T * Hh, Hh, T, 0.0f, nullptr);
    }

    // 4) conv1d(K=3, SAME) as implicit-im2col GEMM: M=8192, K=3072, N=4096; +bias, relu
    gemm_wmma<3, true><<<dim3(U / 256, Bn * T / 64), 256, 0, stream>>>(
        attn, Hh, cwT, 3 * Hh, hbuf, U, 3 * Hh, 0.0f, conv_b);

    // 5) dense: out[8192,1024] = h * dense_w + dense_b (f32 output)
    gemm_wmma<2, false><<<dim3(Hh / 256, Bn * T / 64), 256, 0, stream>>>(
        hbuf, U, dwT, U, out, Hh, U, 0.0f, dense_b);
}
